// MyEmbeddingModel_2851858284657
// MI455X (gfx1250) — matrix operand order repaired and checked by reading, not compiled, not run on hardware
//
#include <hip/hip_runtime.h>

typedef __bf16 bf16;
typedef __attribute__((ext_vector_type(16))) __bf16 v16bf;
typedef __attribute__((ext_vector_type(8)))  float  v8f;

constexpr int   L_ = 2, D_ = 2048, H_ = 16, HD_ = 128, FF_ = 5632, R_ = 16;
constexpr int   B_ = 4, S_ = 512;
constexpr int   MS_ = B_ * S_;              // 2048 token rows
constexpr float LORA_SCALE_ = 1.4f;
constexpr float TEMP_ = 0.05f;

// ---------------------------------------------------------------------------
// f32 -> bf16 elementwise conversion (activations)
// ---------------------------------------------------------------------------
__global__ void cvt_f32_to_bf16(const float* __restrict__ in,
                                bf16* __restrict__ out, long n) {
  long i = (long)blockIdx.x * blockDim.x + threadIdx.x;
  long stride = (long)gridDim.x * blockDim.x;
  for (; i < n; i += stride) out[i] = (bf16)in[i];
}

// ---------------------------------------------------------------------------
// Fused f32->bf16 + transpose (weights, once per launch).  in: rows x cols,
// out: cols x rows.  All dims are multiples of 16, LDS-tiled for coalescing.
// ---------------------------------------------------------------------------
__global__ __launch_bounds__(256) void transpose_f32_bf16_kernel(
    const float* __restrict__ in, bf16* __restrict__ out, int rows, int cols) {
  __shared__ float tile[16][17];
  int c0 = blockIdx.x * 16, r0 = blockIdx.y * 16;
  int tx = threadIdx.x & 15, ty = threadIdx.x >> 4;
  tile[ty][tx] = in[(size_t)(r0 + ty) * cols + (c0 + tx)];
  __syncthreads();
  out[(size_t)(c0 + ty) * rows + (r0 + tx)] = (bf16)tile[tx][ty];
}

// ---------------------------------------------------------------------------
// V head-transpose: vf (B,S,H,HD) f32  ->  vt[z= b*H+h][d][s] bf16
// ---------------------------------------------------------------------------
__global__ __launch_bounds__(256) void v_transpose_kernel(
    const float* __restrict__ v, bf16* __restrict__ vt) {
  __shared__ float tile[16][17];
  int z = blockIdx.z;                      // b*H + h
  int b = z / H_, hh = z % H_;
  int s0 = blockIdx.x * 16, d0 = blockIdx.y * 16;
  int tx = threadIdx.x & 15, ty = threadIdx.x >> 4;
  tile[ty][tx] = v[(((size_t)b * S_ + s0 + ty) * H_ + hh) * HD_ + d0 + tx];
  __syncthreads();
  vt[((size_t)z * HD_ + d0 + ty) * S_ + (s0 + tx)] = (bf16)tile[tx][ty];
}

// ---------------------------------------------------------------------------
// Embedding gather: h[row,:] = emb[ids[row],:]
// ---------------------------------------------------------------------------
__global__ __launch_bounds__(256) void embed_kernel(const int* __restrict__ ids,
                                                    const float* __restrict__ emb,
                                                    float* __restrict__ h) {
  int row = blockIdx.x;
  const float* src = emb + (size_t)ids[row] * D_;
  float* dst = h + (size_t)row * D_;
  for (int d = threadIdx.x; d < D_; d += 256) dst[d] = src[d];
}

// ---------------------------------------------------------------------------
// RMSNorm (f32 in) -> bf16 out, one block per row
// ---------------------------------------------------------------------------
__global__ __launch_bounds__(256) void rmsnorm_bf16_kernel(
    const float* __restrict__ h, const float* __restrict__ w,
    bf16* __restrict__ out) {
  __shared__ float red[8];
  __shared__ float scl;
  int row = blockIdx.x;
  int lane = threadIdx.x & 31, wave = threadIdx.x >> 5;
  const float* x = h + (size_t)row * D_;
  float ss = 0.f;
  for (int d = threadIdx.x; d < D_; d += 256) { float v = x[d]; ss += v * v; }
  for (int o = 16; o > 0; o >>= 1) ss += __shfl_xor(ss, o, 32);
  if (lane == 0) red[wave] = ss;
  __syncthreads();
  if (threadIdx.x == 0) {
    float t = 0.f;
    for (int i = 0; i < 8; i++) t += red[i];
    scl = rsqrtf(t / (float)D_ + 1e-6f);
  }
  __syncthreads();
  float s = scl;
  bf16* o_ = out + (size_t)row * D_;
  for (int d = threadIdx.x; d < D_; d += 256) o_[d] = (bf16)(x[d] * s * w[d]);
}

// ---------------------------------------------------------------------------
// RoPE on (B,S,H,HD) f32 -> bf16, one thread per rotation pair
// ---------------------------------------------------------------------------
__global__ __launch_bounds__(256) void rope_bf16_kernel(
    const float* __restrict__ in, bf16* __restrict__ out) {
  long i = (long)blockIdx.x * blockDim.x + threadIdx.x;
  long total = (long)B_ * S_ * H_ * (HD_ / 2);
  if (i >= total) return;
  int d = (int)(i & 63);
  long rem = i >> 6;
  int hh = (int)(rem % H_);
  long rem2 = rem / H_;
  int s = (int)(rem2 % S_);
  int b = (int)(rem2 / S_);
  float freq = powf(10000.0f, -(float)d * (1.0f / 64.0f));
  float ang = (float)s * freq;
  float sn, cs;
  sincosf(ang, &sn, &cs);
  size_t base = (((size_t)b * S_ + s) * H_ + hh) * HD_;
  float x1 = in[base + d], x2 = in[base + 64 + d];
  out[base + d]      = (bf16)(x1 * cs - x2 * sn);
  out[base + 64 + d] = (bf16)(x1 * sn + x2 * cs);
}

// ---------------------------------------------------------------------------
// Masked causal softmax over S=512 scores per row, one wave per row -> bf16
// ---------------------------------------------------------------------------
__global__ __launch_bounds__(256) void softmax_bf16_kernel(
    const float* __restrict__ scores, const int* __restrict__ mask,
    bf16* __restrict__ attn) {
  int wave = threadIdx.x >> 5, lane = threadIdx.x & 31;
  long rowIdx = (long)blockIdx.x * 8 + wave;      // 0 .. B*H*S-1
  int qi = (int)(rowIdx % S_);
  long bh = rowIdx / S_;
  int b = (int)(bh / H_);
  const float* srow = scores + rowIdx * S_;
  const int* mrow = mask + (size_t)b * S_;
  float vals[16];
  float mx = -1e30f;
#pragma unroll
  for (int j = 0; j < 16; j++) {
    int kk = lane + j * 32;
    bool ok = (kk <= qi) && (mrow[kk] > 0);
    float v = ok ? srow[kk] : -1e9f;
    vals[j] = v;
    mx = fmaxf(mx, v);
  }
  for (int o = 16; o > 0; o >>= 1) mx = fmaxf(mx, __shfl_xor(mx, o, 32));
  float sum = 0.f;
#pragma unroll
  for (int j = 0; j < 16; j++) { vals[j] = __expf(vals[j] - mx); sum += vals[j]; }
  for (int o = 16; o > 0; o >>= 1) sum += __shfl_xor(sum, o, 32);
  float inv = 1.f / sum;
  bf16* arow = attn + rowIdx * S_;
#pragma unroll
  for (int j = 0; j < 16; j++) arow[lane + j * 32] = (bf16)(vals[j] * inv);
}

// ---------------------------------------------------------------------------
// SiLU(g)*u -> bf16
// ---------------------------------------------------------------------------
__global__ void silu_mul_bf16_kernel(const float* __restrict__ g,
                                     const float* __restrict__ u,
                                     bf16* __restrict__ out, long n) {
  long i = (long)blockIdx.x * blockDim.x + threadIdx.x;
  long stride = (long)gridDim.x * blockDim.x;
  for (; i < n; i += stride) {
    float gv = g[i];
    out[i] = (bf16)((gv / (1.f + __expf(-gv))) * u[i]);
  }
}

// ---------------------------------------------------------------------------
// Sequence index selection (matches reference left_pad logic)
// ---------------------------------------------------------------------------
__global__ void seq_idx_kernel(const int* __restrict__ mask, int* __restrict__ idx) {
  int lastTot = 0;
  int sums[B_];
  for (int b = 0; b < B_; b++) {
    int s = 0;
    for (int j = 0; j < S_; j++) s += mask[b * S_ + j];
    sums[b] = s;
    lastTot += mask[b * S_ + S_ - 1];
  }
  bool left_pad = (lastTot == B_);
  for (int b = 0; b < B_; b++) idx[b] = left_pad ? (S_ - 1) : (sums[b] - 1);
}

// ---------------------------------------------------------------------------
// Final rep: rmsnorm(lnf) of selected row then L2-normalize
// ---------------------------------------------------------------------------
__global__ __launch_bounds__(256) void final_rep_kernel(
    const float* __restrict__ h, const float* __restrict__ lnf,
    const int* __restrict__ idx, float* __restrict__ rep) {
  __shared__ float red[8];
  __shared__ float scale0, scale1;
  int b = blockIdx.x;
  int lane = threadIdx.x & 31, wave = threadIdx.x >> 5;
  const float* x = h + ((size_t)b * S_ + idx[b]) * D_;
  float ss = 0.f;
  for (int d = threadIdx.x; d < D_; d += 256) { float v = x[d]; ss += v * v; }
  for (int o = 16; o > 0; o >>= 1) ss += __shfl_xor(ss, o, 32);
  if (lane == 0) red[wave] = ss;
  __syncthreads();
  if (threadIdx.x == 0) {
    float t = 0.f;
    for (int i = 0; i < 8; i++) t += red[i];
    scale0 = rsqrtf(t / (float)D_ + 1e-6f);
  }
  __syncthreads();
  float s0 = scale0;
  float nn = 0.f;
  for (int d = threadIdx.x; d < D_; d += 256) {
    float y = x[d] * s0 * lnf[d];
    nn += y * y;
  }
  for (int o = 16; o > 0; o >>= 1) nn += __shfl_xor(nn, o, 32);
  if (lane == 0) red[wave] = nn;
  __syncthreads();
  if (threadIdx.x == 0) {
    float t = 0.f;
    for (int i = 0; i < 8; i++) t += red[i];
    scale1 = rsqrtf(t);
  }
  __syncthreads();
  float s1 = scale1;
  for (int d = threadIdx.x; d < D_; d += 256)
    rep[(size_t)b * D_ + d] = x[d] * s0 * lnf[d] * s1;
}

// ---------------------------------------------------------------------------
// 4x4 contrastive loss
// ---------------------------------------------------------------------------
__global__ __launch_bounds__(256) void loss_kernel(const float* __restrict__ repT,
                                                   const float* __restrict__ repM,
                                                   float* __restrict__ out) {
  __shared__ float sims[B_][B_];
  int wave = threadIdx.x >> 5, lane = threadIdx.x & 31;
  for (int p = wave * 2; p < wave * 2 + 2; p++) {   // 16 pairs over 8 waves
    int i = p >> 2, j = p & 3;
    float s = 0.f;
    for (int d = lane; d < D_; d += 32) s += repT[i * D_ + d] * repM[j * D_ + d];
    for (int o = 16; o > 0; o >>= 1) s += __shfl_xor(s, o, 32);
    if (lane == 0) sims[i][j] = s / TEMP_;
  }
  __syncthreads();
  if (threadIdx.x == 0) {
    float loss = 0.f;
    for (int i = 0; i < B_; i++) {
      float mx = -1e30f;
      for (int j = 0; j < B_; j++) mx = fmaxf(mx, sims[i][j]);
      float sum = 0.f;
      for (int j = 0; j < B_; j++) sum += expf(sims[i][j] - mx);
      loss -= (sims[i][i] - mx - logf(sum));
    }
    out[0] = loss / (float)B_;
  }
}

// ---------------------------------------------------------------------------
// 32-element (64B) guarded chunk copy global->LDS: 4x b128 loads + 4x b128
// LDS stores.  Guard is per 8-element chunk (all K in this model are
// multiples of 8), so loads stay fully vectorized.
// ---------------------------------------------------------------------------
__device__ __forceinline__ void load_chunk32(const bf16* __restrict__ src,
                                             bf16* __restrict__ dst,
                                             bool rowOK, int kbase, int K) {
  uint4 u[4];
#pragma unroll
  for (int i = 0; i < 4; i++) {
    if (rowOK && (kbase + i * 8) < K) u[i] = *(const uint4*)(src + i * 8);
    else u[i] = make_uint4(0u, 0u, 0u, 0u);
  }
#pragma unroll
  for (int i = 0; i < 4; i++) *(uint4*)(dst + i * 8) = u[i];
}

// ---------------------------------------------------------------------------
// WMMA bf16 TN-GEMM: C[M,N](f32) = alpha * A[M,K] x Bt[N,K]^T (+C if ACCUM)
// Both A and Bt are row-major with K contiguous -> all tile loads are b128.
// 128x256x64 workgroup tile, 8 wave32 waves each owning a 64x64 C tile
// (4x4 fragments, 16 v_wmma_f32_16x16x32_bf16 per 32-wide K-step).
// Batched via z with a 2-level (outer/inner) stride scheme for attention.
// ---------------------------------------------------------------------------
template <bool ACCUM>
__global__ void __launch_bounds__(256) gemm_bf16_tn(
    const bf16* __restrict__ A, long lda, const bf16* __restrict__ Bt, long ldb,
    float* __restrict__ C, long ldc, int M, int N, int K, float alpha,
    int zInner, long sAo, long sAi, long sBo, long sBi, long sCo, long sCi) {
  constexpr int BM = 128, BN = 256, BK = 64, KP = 72;  // KP pad keeps 16B align
  __shared__ bf16 As[BM][KP];
  __shared__ bf16 Bs[BN][KP];

  int z = blockIdx.z;
  int zo = z / zInner, zi = z % zInner;
  A  += (size_t)zo * sAo + (size_t)zi * sAi;
  Bt += (size_t)zo * sBo + (size_t)zi * sBi;
  C  += (size_t)zo * sCo + (size_t)zi * sCi;

  int row0 = blockIdx.y * BM;
  int col0 = blockIdx.x * BN;
  int tid = threadIdx.x;
  int lane = tid & 31;
  int wave = tid >> 5;
  int wm = wave & 1;    // 0..1 -> 64-row slab
  int wn = wave >> 1;   // 0..3 -> 64-col slab
  int lh = lane >> 4;   // lane half (ISA K-half select)
  int ll = lane & 15;

  v8f acc[4][4];
#pragma unroll
  for (int mt = 0; mt < 4; mt++)
#pragma unroll
    for (int nt = 0; nt < 4; nt++)
#pragma unroll
      for (int r = 0; r < 8; r++) acc[mt][nt][r] = 0.f;

  int lm = tid >> 1;            // 0..127
  int lk = (tid & 1) * 32;      // 0 or 32

  for (int k0 = 0; k0 < K; k0 += BK) {
    if (k0 + BK < K) {  // global_prefetch_b8 next tiles
      __builtin_prefetch(A  + (size_t)(row0 + lm) * lda + k0 + BK, 0, 1);
      __builtin_prefetch(Bt + (size_t)(col0 + lm) * ldb + k0 + BK, 0, 1);
    }
    // A tile: 128 rows x 64 k (2 threads/row x 32 elems)
    load_chunk32(A + (size_t)(row0 + lm) * lda + k0 + lk, &As[lm][lk],
                 (row0 + lm) < M, k0 + lk, K);
    // B tile: 256 rows x 64 k (2 rows per thread)
#pragma unroll
    for (int r = 0; r < 2; r++) {
      int n = lm + r * 128;
      load_chunk32(Bt + (size_t)(col0 + n) * ldb + k0 + lk, &Bs[n][lk],
                   (col0 + n) < N, k0 + lk, K);
    }
    __syncthreads();

#pragma unroll
    for (int kk2 = 0; kk2 < BK; kk2 += 32) {
      // fragments in contiguous k order k = 16h+s, same for A and B (A 16x32, B 32x16)
      v16bf afrag[4], bfrag[4];
#pragma unroll
      for (int mt = 0; mt < 4; mt++) {
        int m = wm * 64 + mt * 16 + ll;
#pragma unroll
        for (int e = 0; e < 8; e++) {
          afrag[mt][e]     = As[m][kk2 + 16 * lh + e];      // K = 16h+e
          afrag[mt][8 + e] = As[m][kk2 + 16 * lh + 8 + e];  // K = 16h+8+e
        }
      }
#pragma unroll
      for (int nt = 0; nt < 4; nt++) {
        int n = wn * 64 + nt * 16 + ll;
#pragma unroll
        for (int e = 0; e < 16; e++)
          bfrag[nt][e] = Bs[n][kk2 + 16 * lh + e];          // K = 16h+e
      }
#pragma unroll
      for (int mt = 0; mt < 4; mt++)
#pragma unroll
        for (int nt = 0; nt < 4; nt++)
          acc[mt][nt] = __builtin_amdgcn_wmma_f32_16x16x32_bf16(
              false, afrag[mt], false, bfrag[nt], (short)0, acc[mt][nt],
              false, false);
    }
    __syncthreads();
  }

  // store per ISA f32 C layout: VGPR r -> row r+8h, col = lane&15
#pragma unroll
  for (int mt = 0; mt < 4; mt++) {
#pragma unroll
    for (int nt = 0; nt < 4; nt++) {
      int col = col0 + wn * 64 + nt * 16 + ll;
      if (col >= N) continue;
#pragma unroll
      for (int r = 0; r < 8; r++) {
        int row = row0 + wm * 64 + mt * 16 + r + 8 * lh;
        if (row >= M) continue;
        size_t off = (size_t)row * ldc + col;
        float v = alpha * acc[mt][nt][r];
        C[off] = ACCUM ? (C[off] + v) : v;
      }
    }
  }
}

// ---------------------------------------------------------------------------
extern "C" void kernel_launch(void* const* d_in, const int* in_sizes, int n_in,
                              void* d_out, int out_size, void* d_ws, size_t ws_size,
                              hipStream_t stream) {
  (void)in_sizes; (void)n_in; (void)out_size; (void)ws_size;

  const int*   ids_t  = (const int*)d_in[0];
  const int*   mask_t = (const int*)d_in[1];
  const int*   ids_m  = (const int*)d_in[2];
  const int*   mask_m = (const int*)d_in[3];
  const float* emb    = (const float*)d_in[4];
  const float* ln1    = (const float*)d_in[5];
  const float* ln2    = (const float*)d_in[6];
  const float* lnf    = (const float*)d_in[7];

  // ---- workspace bump allocator ----
  char* basep = (char*)d_ws;
  size_t off = 0;
  auto alloc = [&](size_t bytes) -> void* {
    void* p = basep + off;
    off = (off + bytes + 255) & ~(size_t)255;
    return p;
  };

  static const int dins[7]  = {D_, D_, D_, D_, D_, D_, FF_};
  static const int douts[7] = {D_, D_, D_, D_, FF_, FF_, D_};
  // Transposed bf16 weights: Wt = N x K, At = R x K, Bt = N x R  (per layer)
  bf16 *Wt[7], *At[7], *Bt[7];
  for (int n = 0; n < 7; n++) {
    Wt[n] = (bf16*)alloc((size_t)L_ * dins[n] * douts[n] * sizeof(bf16));
    At[n] = (bf16*)alloc((size_t)L_ * dins[n] * R_ * sizeof(bf16));
    Bt[n] = (bf16*)alloc((size_t)L_ * R_ * douts[n] * sizeof(bf16));
  }
  float* h    = (float*)alloc((size_t)MS_ * D_ * 4);
  bf16*  xbf  = (bf16*) alloc((size_t)MS_ * D_ * 2);
  float* qf   = (float*)alloc((size_t)MS_ * D_ * 4);
  float* kf   = (float*)alloc((size_t)MS_ * D_ * 4);
  float* vf   = (float*)alloc((size_t)MS_ * D_ * 4);
  bf16*  qbf  = (bf16*) alloc((size_t)MS_ * D_ * 2);
  bf16*  kbf  = (bf16*) alloc((size_t)MS_ * D_ * 2);
  bf16*  vtbf = (bf16*) alloc((size_t)MS_ * D_ * 2);           // [b,h][d][s]
  float* sc   = (float*)alloc((size_t)B_ * H_ * S_ * S_ * 4);  // scores; alias gf
  bf16*  attn = (bf16*) alloc((size_t)B_ * H_ * S_ * S_ * 2);  // alias mbf
  float* of   = (float*)alloc((size_t)MS_ * D_ * 4);
  bf16*  obf  = (bf16*) alloc((size_t)MS_ * D_ * 2);
  float* uf   = (float*)alloc((size_t)MS_ * FF_ * 4);
  float* tmpl = (float*)alloc((size_t)MS_ * R_ * 4);
  bf16*  tmplb= (bf16*) alloc((size_t)MS_ * R_ * 2);
  float* repT = (float*)alloc((size_t)B_ * D_ * 4);
  float* repM = (float*)alloc((size_t)B_ * D_ * 4);
  int*   idx  = (int*)  alloc(256);
  float* gf  = sc;    // scores dead after softmax; reuse for MLP gate
  bf16*  mbf = attn;  // attn dead after P*V; reuse for silu*u

  // ---- weights: f32 -> bf16 with transpose to NxK (once per launch) ----
  for (int n = 0; n < 7; n++) {
    const float* W = (const float*)d_in[8 + 3 * n];
    const float* A = (const float*)d_in[9 + 3 * n];
    const float* Bm = (const float*)d_in[10 + 3 * n];
    int Kd = dins[n], Nd = douts[n];
    for (int l = 0; l < L_; l++) {
      transpose_f32_bf16_kernel<<<dim3(Nd / 16, Kd / 16), 256, 0, stream>>>(
          W + (size_t)l * Kd * Nd, Wt[n] + (size_t)l * Nd * Kd, Kd, Nd);
      transpose_f32_bf16_kernel<<<dim3(R_ / 16, Kd / 16), 256, 0, stream>>>(
          A + (size_t)l * Kd * R_, At[n] + (size_t)l * R_ * Kd, Kd, R_);
      transpose_f32_bf16_kernel<<<dim3(Nd / 16, R_ / 16), 256, 0, stream>>>(
          Bm + (size_t)l * R_ * Nd, Bt[n] + (size_t)l * Nd * R_, R_, Nd);
    }
  }

  auto gemm = [&](const bf16* A, long lda, const bf16* Bm, long ldb, float* C,
                  long ldc, int M, int N, int K, float alpha, bool accum,
                  int batch, int zInner, long sAo, long sAi, long sBo,
                  long sBi, long sCo, long sCi) {
    dim3 grid((N + 255) / 256, (M + 127) / 128, batch);
    dim3 blk(256);
    if (accum)
      gemm_bf16_tn<true><<<grid, blk, 0, stream>>>(A, lda, Bm, ldb, C, ldc, M, N, K, alpha, zInner, sAo, sAi, sBo, sBi, sCo, sCi);
    else
      gemm_bf16_tn<false><<<grid, blk, 0, stream>>>(A, lda, Bm, ldb, C, ldc, M, N, K, alpha, zInner, sAo, sAi, sBo, sBi, sCo, sCi);
  };

  // y = x@W (+y if accum)  then  y += 1.4 * (x@A)@B   (all weights NxK form)
  auto proj = [&](const bf16* x, int n, int l, float* out, bool accum) {
    int Kd = dins[n], Nd = douts[n];
    const bf16* W = Wt[n] + (size_t)l * Nd * Kd;
    const bf16* Aa = At[n] + (size_t)l * R_ * Kd;
    const bf16* Bm = Bt[n] + (size_t)l * Nd * R_;
    gemm(x, Kd, W, Kd, out, Nd, MS_, Nd, Kd, 1.f, accum, 1, 1, 0, 0, 0, 0, 0, 0);
    gemm(x, Kd, Aa, Kd, tmpl, R_, MS_, R_, Kd, 1.f, false, 1, 1, 0, 0, 0, 0, 0, 0);
    cvt_f32_to_bf16<<<64, 256, 0, stream>>>(tmpl, tmplb, (long)MS_ * R_);
    gemm(tmplb, R_, Bm, R_, out, Nd, MS_, Nd, R_, LORA_SCALE_, true, 1, 1, 0, 0, 0, 0, 0, 0);
  };

  auto encode = [&](const int* ids, const int* mask, float* rep) {
    embed_kernel<<<MS_, 256, 0, stream>>>(ids, emb, h);
    for (int l = 0; l < L_; l++) {
      rmsnorm_bf16_kernel<<<MS_, 256, 0, stream>>>(h, ln1 + (size_t)l * D_, xbf);
      proj(xbf, 0, l, qf, false);
      proj(xbf, 1, l, kf, false);
      proj(xbf, 2, l, vf, false);
      rope_bf16_kernel<<<8192, 256, 0, stream>>>(qf, qbf);
      rope_bf16_kernel<<<8192, 256, 0, stream>>>(kf, kbf);
      v_transpose_kernel<<<dim3(S_ / 16, HD_ / 16, B_ * H_), 256, 0, stream>>>(vf, vtbf);
      // scores[b,h] = Q(512x128) * K^T / sqrt(HD), batched over B*H
      gemm(qbf, D_, kbf, D_, sc, S_, S_, S_, HD_, 0.08838834764831845f, false,
           B_ * H_, H_, (long)S_ * D_, HD_, (long)S_ * D_, HD_,
           (long)H_ * S_ * S_, (long)S_ * S_);
      softmax_bf16_kernel<<<B_ * H_ * S_ / 8, 256, 0, stream>>>(sc, mask, attn);
      // o[b,h] = P(512x512) * V(512x128); V supplied head-transposed (HDxS)
      gemm(attn, S_, vtbf, S_, of, D_, S_, HD_, S_, 1.f, false, B_ * H_, H_,
           (long)H_ * S_ * S_, (long)S_ * S_, (long)H_ * HD_ * S_,
           (long)HD_ * S_, (long)S_ * D_, HD_);
      cvt_f32_to_bf16<<<2048, 256, 0, stream>>>(of, obf, (long)MS_ * D_);
      proj(obf, 3, l, h, true);   // h += lora(o)
      rmsnorm_bf16_kernel<<<MS_, 256, 0, stream>>>(h, ln2 + (size_t)l * D_, xbf);
      proj(xbf, 4, l, gf, false);
      proj(xbf, 5, l, uf, false);
      silu_mul_bf16_kernel<<<4096, 256, 0, stream>>>(gf, uf, mbf, (long)MS_ * FF_);
      proj(mbf, 6, l, h, true);   // h += lora(silu(g)*u)
    }
    seq_idx_kernel<<<1, 1, 0, stream>>>(mask, idx);
    final_rep_kernel<<<B_, 256, 0, stream>>>(h, lnf, idx, rep);
  };

  encode(ids_t, mask_t, repT);
  encode(ids_m, mask_m, repM);
  loss_kernel<<<1, 256, 0, stream>>>(repT, repM, (float*)d_out);
}
